// MTL_Classifier_52621939310866
// MI455X (gfx1250) — matrix operand order repaired and checked by reading, not compiled, not run on hardware
//
#include <hip/hip_runtime.h>
#include <hip/hip_bf16.h>

// ---------------------------------------------------------------------------
// BiLSTM(x2) + linear emissions + CRF NLL for MI455X (gfx1250, wave32, WMMA).
// Phase 1: bf16 conversion + weight transposes.
// Phase 2: time-parallel input projections as bf16 WMMA GEMMs (f32 accum),
//          A-tiles staged with GLOBAL_LOAD_ASYNC_TO_LDS_B128 (ASYNCcnt).
// Phase 3: 4 recurrences; each direction split over 8 WGs with whh.T slice
//          resident in LDS (64KB, fragment-packed, B-fragments hoisted into
//          registers for all 1024 steps), per-step 32x1024x256 gate GEMM via
//          WMMA, cross-WG h exchange through L2 + global split barrier with
//          async LDS re-gather of the hidden state.
// Phase 4: emission GEMMs (N=16) + sequential CRF forward/logsumexp.
// ---------------------------------------------------------------------------

#define BATCH  32
#define SEQ    1024
#define EMB    512
#define HID    256
#define BS     (BATCH * SEQ)      // 32768 rows
#define GATES  1024               // 4*HID
#define PAD_T  0
#define SOS_T  1
#define EOS_T  2

typedef __attribute__((ext_vector_type(16))) __bf16 v16bf;
typedef __attribute__((ext_vector_type(8)))  __bf16 v8bf;
typedef __attribute__((ext_vector_type(8)))  float  v8f;

// ---------------- CDNA5 async global->LDS copy (ISA 15.18.3, op 98) ---------
// Per-lane: LDS[lds_off] = MEM[gaddr], 16 bytes, tracked by ASYNCcnt.
__device__ __forceinline__ void async_copy16(unsigned lds_off, const void* gptr) {
  asm volatile("global_load_async_to_lds_b128 %0, %1, off"
               :: "v"(lds_off), "v"((unsigned long long)(size_t)gptr)
               : "memory");
}
__device__ __forceinline__ void async_wait0() {
  asm volatile("s_wait_asynccnt 0" ::: "memory");
}

// ---------------- WMMA fragment helpers (ISA 7.12.2 layouts) ----------------

// A fragment: 16x32 bf16 tile, row-major source with row stride `stride`.
// lane<16 holds row=lane, K in {0..7,16..23}; lane>=16 same row, K+8.
// Per lane this is two contiguous 16B runs -> compiler emits ds_load_b128 x2.
__device__ __forceinline__ v16bf load_a_frag(const __bf16* base, int stride) {
  const int lane = threadIdx.x & 31;
  const int row  = lane & 15;
  const int kb   = (lane & 16) ? 8 : 0;
  v16bf a;
#pragma unroll
  for (int v = 0; v < 8; ++v) {
    const int k0 = ((v < 4) ? (2 * v) : (16 + 2 * (v - 4))) + kb;
    a[2 * v]     = base[row * stride + k0];
    a[2 * v + 1] = base[row * stride + k0 + 1];
  }
  return a;
}

// B fragment from fragment-packed LDS: [32 lanes][16 bf16] contiguous.
__device__ __forceinline__ v16bf load_frag_packed(const __bf16* base) {
  const int lane = threadIdx.x & 31;
  return *(const v16bf*)(base + lane * 16);
}

// Packed-store index for B element (k in [0,32), n in [0,16)) of one tile:
//   lane = n + 16*((k>>3)&1), slot = (k&7) + 8*(k>>4)  ->  offset = lane*16 + slot
__device__ __forceinline__ int b_pack_idx(int k, int n) {
  return ((n + (((k >> 3) & 1) << 4)) << 4) + ((k & 7) + ((k >> 4) << 3));
}

// ---------------- prep kernels ----------------

__global__ void k_f2bf(const float* __restrict__ s, __bf16* __restrict__ d, size_t n) {
  size_t i = (size_t)blockIdx.x * blockDim.x + threadIdx.x;
  const size_t st = (size_t)gridDim.x * blockDim.x;
  for (; i < n; i += st) d[i] = (__bf16)s[i];
}

// d[c*R + r] = s[r*C + c]   (R x C f32  ->  C x R bf16)
__global__ void k_transpose_bf(const float* __restrict__ s, __bf16* __restrict__ d,
                               int R, int C) {
  size_t i = (size_t)blockIdx.x * blockDim.x + threadIdx.x;
  const size_t n = (size_t)R * C;
  const size_t st = (size_t)gridDim.x * blockDim.x;
  for (; i < n; i += st) {
    int r = (int)(i / C), c = (int)(i % C);
    d[(size_t)c * R + r] = (__bf16)s[(size_t)r * C + c];
  }
}

// d[c*ldd + r] = (r<R) ? s[r*C+c] : 0     (zero-padded transpose)
__global__ void k_transpose_pad_bf(const float* __restrict__ s, __bf16* __restrict__ d,
                                   int R, int C, int ldd) {
  size_t i = (size_t)blockIdx.x * blockDim.x + threadIdx.x;
  const size_t n = (size_t)C * ldd;
  const size_t st = (size_t)gridDim.x * blockDim.x;
  for (; i < n; i += st) {
    int c = (int)(i / ldd), r = (int)(i % ldd);
    d[(size_t)c * ldd + r] = (r < R) ? (__bf16)s[(size_t)r * C + c] : (__bf16)0.0f;
  }
}

__global__ void k_bias_add(const float* __restrict__ a, const float* __restrict__ b,
                           float* __restrict__ o, int n) {
  int i = blockIdx.x * blockDim.x + threadIdx.x;
  if (i < n) o[i] = a[i] + b[i];
}

__global__ void k_copy_pad(const float* __restrict__ s, float* __restrict__ d,
                           int n, int total) {
  int i = blockIdx.x * blockDim.x + threadIdx.x;
  if (i < total) d[i] = (i < n) ? s[i] : 0.0f;
}

__global__ void k_zero_u32(unsigned* __restrict__ p, int n) {
  int i = blockIdx.x * blockDim.x + threadIdx.x;
  if (i < n) p[i] = 0u;
}

// ---------------- generic bf16 WMMA GEMM:  C = A(bf16) * B(bf16) + bias ----
// Tile: 128(M) x 128(N) per WG, BK=32. 256 threads = 8 waves; wave w owns
// N-tile w and all 8 M-tiles. Requires M%128==0, K%32==0, rows 16B aligned
// (true for all call sites); N may be ragged (guarded).
// sA staged via async global->LDS b128; sB fragment-packed via registers.

__global__ void __launch_bounds__(256) k_gemm_bf16(
    const __bf16* __restrict__ A, long lda,
    const __bf16* __restrict__ Bm, long ldb,
    const float* __restrict__ bias,
    float* __restrict__ C, long ldc,
    int M, int N, int K) {
  extern __shared__ __align__(64) char smem[];
  __bf16* sA = (__bf16*)smem;                  // 128 x 32 (8KB)
  __bf16* sB = (__bf16*)(smem + 128 * 32 * 2); // 8 x 512 packed (8KB)

  const int tid  = threadIdx.x;
  const int lane = tid & 31;
  const int wave = tid >> 5;
  const int m0 = blockIdx.x * 128;
  const int n0 = blockIdx.y * 128;
  const int nt = wave;
  const unsigned ldsA = __builtin_amdgcn_groupstaticsize();  // dyn-LDS base

  v8f acc[8] = {{}, {}, {}, {}, {}, {}, {}, {}};

  for (int kk = 0; kk < K; kk += 32) {
    __syncthreads();
    // A stage: 128 rows x 32 cols = 512 16B chunks, async global->LDS.
#pragma unroll
    for (int ch = tid; ch < 512; ch += 256) {
      const int r = ch >> 2, c8 = ch & 3;
      async_copy16(ldsA + (unsigned)(r * 32 + c8 * 8) * 2,
                   &A[(size_t)(m0 + r) * lda + kk + c8 * 8]);
    }
    // B stage: 32 k-rows x 128 cols = 512 8-element chunks; vector load from
    // global (guarded on N), scatter-store into fragment-packed layout.
#pragma unroll
    for (int ch = tid; ch < 512; ch += 256) {
      const int r = ch >> 4, c8 = ch & 15;    // k-row, 8-col chunk
      const int gn0 = n0 + c8 * 8;
      v8bf val;
#pragma unroll
      for (int e = 0; e < 8; ++e) val[e] = (__bf16)0.0f;
      if (gn0 + 8 <= N)
        val = *(const v8bf*)&Bm[(size_t)(kk + r) * ldb + gn0];
#pragma unroll
      for (int e = 0; e < 8; ++e) {
        const int nl = c8 * 8 + e;            // 0..127 local col
        const int tile = nl >> 4, nn = nl & 15;
        sB[tile * 512 + b_pack_idx(r, nn)] = val[e];
      }
    }
    if (kk + 32 < K) {  // hint next K-slice (global_prefetch_b8)
      __builtin_prefetch(&A[(size_t)(m0 + (tid & 127)) * lda + kk + 32], 0, 1);
    }
    async_wait0();
    __syncthreads();
    // Load B frag + all 8 A frags into distinct regs, then 8 back-to-back WMMAs.
    v16bf bf = load_frag_packed(sB + nt * 512);
    v16bf af[8];
#pragma unroll
    for (int mt = 0; mt < 8; ++mt) af[mt] = load_a_frag(sA + mt * 16 * 32, 32);
#pragma unroll
    for (int mt = 0; mt < 8; ++mt) {
      acc[mt] = __builtin_amdgcn_wmma_f32_16x16x32_bf16(
          false, af[mt], false, bf, (short)0, acc[mt], false, false);
    }
  }

  const int n  = lane & 15;
  const int mo = (lane & 16) ? 8 : 0;
  const int gn = n0 + nt * 16 + n;
  if (gn < N) {
    const float bv = bias ? bias[gn] : 0.0f;
#pragma unroll
    for (int mt = 0; mt < 8; ++mt) {
#pragma unroll
      for (int v = 0; v < 8; ++v) {
        int gm = m0 + mt * 16 + mo + v;
        if (gm < M) C[(size_t)gm * ldc + gn] = acc[mt][v] + bv;
      }
    }
  }
}

// ---------------- recurrent LSTM scan ----------------
// 32 WGs: dir = blockIdx.x>>3 (0 main-fwd, 1 main-bwd, 2 bin-fwd, 3 bin-bwd),
// part = blockIdx.x&7 owns hidden units [part*32, part*32+32).
// LDS: whh.T slice 64KB fragment-packed [kk:8][nt:8][lane:32][16 bf16],
//      full h 32x256 bf16, gate accum 32x128 f32, cell slice 32x32 f32.
// The 8 B-fragments per wave are time-invariant -> hoisted into registers.

__global__ void __launch_bounds__(256) k_recurrent(
    const float* __restrict__ Zin,     // [4][BS][GATES] f32
    const __bf16* __restrict__ Whht,   // [4][HID][GATES] bf16
    const float* __restrict__ h0_main, const float* __restrict__ c0_main,
    const float* __restrict__ h0_bin,  const float* __restrict__ c0_bin,
    __bf16* __restrict__ finalbf,      // [BATCH][SEQ][GATES] bf16
    __bf16* __restrict__ hbuf,         // [4][2][BATCH][HID] bf16
    unsigned* __restrict__ ctr) {      // [4] monotonic barrier counters
  extern __shared__ __align__(64) char smem[];
  __bf16* sW = (__bf16*)smem;                                   // packed 64KB
  __bf16* sH = (__bf16*)(smem + 256 * 128 * 2);                 // [32][256]
  float*  sG = (float*)(smem + 256 * 128 * 2 + 32 * 256 * 2);   // [32][128]
  float*  sC = sG + 32 * 128;                                   // [32][32]

  const int tid  = threadIdx.x;
  const int lane = tid & 31;
  const int wave = tid >> 5;
  const int d    = blockIdx.x >> 3;
  const int part = blockIdx.x & 7;
  const int U0   = part * 32;
  const bool rev = (d & 1);
  const int colOff = d * 256;
  const unsigned sH_lds = __builtin_amdgcn_groupstaticsize() + 256 * 128 * 2;

  const float*  Z  = Zin  + (size_t)d * BS * GATES;
  const __bf16* W  = Whht + (size_t)d * HID * GATES;
  const float*  h0 = (d < 2) ? (h0_main + (size_t)(d & 1) * BATCH * HID)
                             : (h0_bin  + (size_t)(d & 1) * BATCH * HID);
  const float*  c0 = (d < 2) ? (c0_main + (size_t)(d & 1) * BATCH * HID)
                             : (c0_bin  + (size_t)(d & 1) * BATCH * HID);

  // Stage whh.T slice once, fragment-packed. Local col c = g*32+u maps to
  // global gate col g*256+U0+u; (k,c) -> frag (kk=k>>5, nt=c>>4).
  for (int i = tid; i < 256 * 128; i += 256) {
    const int k = i >> 7, c = i & 127;
    const int g = c >> 5, u = c & 31;
    const __bf16 w = W[(size_t)k * GATES + g * 256 + U0 + u];
    const int kk = k >> 5, kl = k & 31;
    const int nt = c >> 4, nn = c & 15;
    sW[(kk * 8 + nt) * 512 + b_pack_idx(kl, nn)] = w;
  }
  for (int i = tid; i < 32 * 256; i += 256) sH[i] = (__bf16)h0[i];
  for (int i = tid; i < 32 * 32; i += 256) {
    int b = i >> 5, u = i & 31;
    sC[i] = c0[b * HID + U0 + u];
  }
  __syncthreads();

  // Hoist this wave's 8 time-invariant B fragments into registers (64 VGPRs).
  v16bf bw[8];
#pragma unroll
  for (int kk = 0; kk < 8; ++kk)
    bw[kk] = load_frag_packed(sW + (kk * 8 + wave) * 512);

  for (int s = 0; s < SEQ; ++s) {
    const int t = rev ? (SEQ - 1 - s) : s;

    // gates(local 128 cols) = h(32x256) @ whh.T-slice(256x128)
    v8f acc0 = {}; v8f acc1 = {};
    const int nt = wave;
#pragma unroll
    for (int kk = 0; kk < 8; ++kk) {
      v16bf a0 = load_a_frag(sH + 0  * 256 + kk * 32, 256);
      v16bf a1 = load_a_frag(sH + 16 * 256 + kk * 32, 256);
      acc0 = __builtin_amdgcn_wmma_f32_16x16x32_bf16(false, a0, false, bw[kk], (short)0, acc0, false, false);
      acc1 = __builtin_amdgcn_wmma_f32_16x16x32_bf16(false, a1, false, bw[kk], (short)0, acc1, false, false);
    }
    {
      const int n  = lane & 15;
      const int mo = (lane & 16) ? 8 : 0;
#pragma unroll
      for (int v = 0; v < 8; ++v) {
        sG[(mo + v)      * 128 + nt * 16 + n] = acc0[v];
        sG[(16 + mo + v) * 128 + nt * 16 + n] = acc1[v];
      }
    }
    __syncthreads();

    // activations + state update for the 32x32 (b,u) slice this WG owns
    __bf16* hn = hbuf + ((size_t)d * 2 + (s & 1)) * (BATCH * HID);
#pragma unroll
    for (int r = 0; r < 4; ++r) {
      const int id = tid + r * 256;
      const int b = id >> 5, u = id & 31;
      const float* zrow = Z + ((size_t)b * SEQ + t) * GATES;
      const float zi = zrow[0 * 256 + U0 + u] + sG[b * 128 + 0 * 32 + u];
      const float zf = zrow[1 * 256 + U0 + u] + sG[b * 128 + 1 * 32 + u];
      const float zg = zrow[2 * 256 + U0 + u] + sG[b * 128 + 2 * 32 + u];
      const float zo = zrow[3 * 256 + U0 + u] + sG[b * 128 + 3 * 32 + u];
      const float iv = 1.0f / (1.0f + __expf(-zi));
      const float fv = 1.0f / (1.0f + __expf(-zf));
      const float gv = tanhf(zg);
      const float ov = 1.0f / (1.0f + __expf(-zo));
      const float cv = fv * sC[b * 32 + u] + iv * gv;
      sC[b * 32 + u] = cv;
      const float hv = ov * tanhf(cv);
      const __bf16 hb = (__bf16)hv;
      finalbf[((size_t)b * SEQ + t) * GATES + colOff + U0 + u] = hb;
      hn[b * HID + U0 + u] = hb;
    }

    // split-barrier across the 8 parts of this direction (monotonic counter)
    __threadfence();
    __syncthreads();
    if (tid == 0) {
      atomicAdd(&ctr[d], 1u);
      const unsigned target = (unsigned)(s + 1) * 8u;
      while (atomicAdd(&ctr[d], 0u) < target) __builtin_amdgcn_s_sleep(2);
    }
    __syncthreads();
    __threadfence();

    if (s + 1 < SEQ) {  // async re-gather of full h (16KB) straight into LDS
#pragma unroll
      for (int ch = tid; ch < 1024; ch += 256)
        async_copy16(sH_lds + (unsigned)ch * 16, hn + ch * 8);
      async_wait0();
    }
    __syncthreads();
  }
}

// ---------------- CRF NLL (forward algorithm + gold-path score) --------------

__global__ void __launch_bounds__(512) k_crf(
    const float* __restrict__ emis, int ldE,
    const float* __restrict__ trans, int NTc,
    const int* __restrict__ tags,
    const float* __restrict__ mask,
    float* __restrict__ out, int outIdx) {
  __shared__ float sT[16 * 16];
  __shared__ float sA[32 * 16];
  __shared__ float sB2[32 * 16];
  __shared__ float sLogZ[32];
  __shared__ float sScore[32];

  const int tid = threadIdx.x;
  const int b = tid >> 4;
  const int j = tid & 15;

  if (tid < 256) {
    int i = tid >> 4, jj = tid & 15;
    sT[tid] = (i < NTc && jj < NTc) ? trans[i * NTc + jj] : -1.0e6f;
  }
  __syncthreads();

  {
    float v = -1.0e30f;
    if (j < NTc) v = sT[SOS_T * 16 + j] + emis[((size_t)b * SEQ) * ldE + j];
    sA[tid] = v;
  }
  __syncthreads();

  float* Acur = sA;
  float* Anew = sB2;
  for (int t = 1; t < SEQ; ++t) {
    const float m = mask[(size_t)b * SEQ + t];
    float res = Acur[b * 16 + j];
    if (j < NTc) {
      float mx = -1.0e30f;
      for (int i = 0; i < NTc; ++i)
        mx = fmaxf(mx, Acur[b * 16 + i] + sT[i * 16 + j]);
      float ssum = 0.0f;
      for (int i = 0; i < NTc; ++i)
        ssum += __expf(Acur[b * 16 + i] + sT[i * 16 + j] - mx);
      const float lse = mx + __logf(ssum);
      const float nv = emis[((size_t)b * SEQ + t) * ldE + j] + lse;
      res = m * nv + (1.0f - m) * res;
    }
    Anew[b * 16 + j] = res;
    __syncthreads();
    float* tmp = Acur; Acur = Anew; Anew = tmp;
  }

  if (j == 0) {
    float mx = -1.0e30f;
    for (int q = 0; q < NTc; ++q)
      mx = fmaxf(mx, Acur[b * 16 + q] + sT[q * 16 + EOS_T]);
    float ssum = 0.0f;
    for (int q = 0; q < NTc; ++q)
      ssum += __expf(Acur[b * 16 + q] + sT[q * 16 + EOS_T] - mx);
    sLogZ[b] = mx + __logf(ssum);
  }

  if (tid < 32) {
    const int bb = tid;
    const int* tg = tags + (size_t)bb * SEQ;
    const float* mk = mask + (size_t)bb * SEQ;
    int prev = tg[0];
    float sc = sT[SOS_T * 16 + prev] + emis[((size_t)bb * SEQ) * ldE + prev];
    int cnt = (mk[0] > 0.0f) ? 1 : 0;
    for (int t = 1; t < SEQ; ++t) {
      const float m = mk[t];
      const int tt = tg[t];
      sc += m * (emis[((size_t)bb * SEQ + t) * ldE + tt] + sT[prev * 16 + tt]);
      prev = tt;
      cnt += (m > 0.0f) ? 1 : 0;
    }
    sc += sT[tg[cnt - 1] * 16 + EOS_T];
    sScore[bb] = sc;
  }
  __syncthreads();
  if (tid == 0) {
    float nll = 0.0f;
    for (int bb = 0; bb < 32; ++bb) nll += sLogZ[bb] - sScore[bb];
    out[outIdx] = nll;
  }
}

// ---------------- workspace layout (bytes; each size is 256B-aligned) -------

static constexpr size_t O_XBF   = 0;                                          // x bf16
static constexpr size_t O_XBBF  = O_XBF   + (size_t)BS * EMB * 2;             // x_binary bf16
static constexpr size_t O_W1F_T = O_XBBF  + (size_t)BS * 3 * EMB * 2;
static constexpr size_t O_W1B_T = O_W1F_T + (size_t)EMB * GATES * 2;
static constexpr size_t O_W2F_T = O_W1B_T + (size_t)EMB * GATES * 2;
static constexpr size_t O_W2B_T = O_W2F_T + (size_t)3 * EMB * GATES * 2;
static constexpr size_t O_WHHT  = O_W2B_T + (size_t)3 * EMB * GATES * 2;      // 4 x [HID][GATES]
static constexpr size_t O_BIAS4 = O_WHHT  + (size_t)4 * HID * GATES * 2;      // 4 x [GATES] f32
static constexpr size_t O_WTAGT = O_BIAS4 + (size_t)4 * GATES * 4;            // [1024][16] bf16
static constexpr size_t O_BTAG  = O_WTAGT + (size_t)GATES * 16 * 2;
static constexpr size_t O_WBINT = O_BTAG  + 256;                              // [512][16] bf16
static constexpr size_t O_BBIN  = O_WBINT + (size_t)(2 * HID) * 16 * 2;
static constexpr size_t O_ZIN   = O_BBIN  + 256;                              // 4 x [BS][GATES] f32
static constexpr size_t O_FINAL = O_ZIN   + (size_t)4 * BS * GATES * 4;       // [BS][GATES] bf16
static constexpr size_t O_EMIS  = O_FINAL + (size_t)BS * GATES * 2;           // [BS][16] f32
static constexpr size_t O_EMISB = O_EMIS  + (size_t)BS * 16 * 4;              // [BS][16] f32
static constexpr size_t O_HBUF  = O_EMISB + (size_t)BS * 16 * 4;              // 4x2x[B][H] bf16
static constexpr size_t O_CTR   = O_HBUF  + (size_t)4 * 2 * BATCH * HID * 2;  // counters

extern "C" void kernel_launch(void* const* d_in, const int* in_sizes, int n_in,
                              void* d_out, int out_size, void* d_ws, size_t ws_size,
                              hipStream_t stream) {
  (void)in_sizes; (void)n_in; (void)out_size; (void)ws_size;

  const float* x      = (const float*)d_in[0];
  const float* xb     = (const float*)d_in[1];
  const float* mask   = (const float*)d_in[2];
  const int*   tags   = (const int*)d_in[3];
  const int*   tagsb  = (const int*)d_in[4];
  const float* trans  = (const float*)d_in[5];
  const float* transb = (const float*)d_in[6];
  const float* w2f_ih = (const float*)d_in[7];
  const float* w2f_hh = (const float*)d_in[8];
  const float* b2f_ih = (const float*)d_in[9];
  const float* b2f_hh = (const float*)d_in[10];
  const float* w2b_ih = (const float*)d_in[11];
  const float* w2b_hh = (const float*)d_in[12];
  const float* b2b_ih = (const float*)d_in[13];
  const float* b2b_hh = (const float*)d_in[14];
  const float* w_bin  = (const float*)d_in[15];
  const float* bb_bin = (const float*)d_in[16];
  const float* w1f_ih = (const float*)d_in[17];
  const float* w1f_hh = (const float*)d_in[18];
  const float* b1f_ih = (const float*)d_in[19];
  const float* b1f_hh = (const float*)d_in[20];
  const float* w1b_ih = (const float*)d_in[21];
  const float* w1b_hh = (const float*)d_in[22];
  const float* b1b_ih = (const float*)d_in[23];
  const float* b1b_hh = (const float*)d_in[24];
  const float* w_tag  = (const float*)d_in[25];
  const float* b_tag  = (const float*)d_in[26];
  const float* h0_bin  = (const float*)d_in[27];
  const float* c0_bin  = (const float*)d_in[28];
  const float* h0_main = (const float*)d_in[29];
  const float* c0_main = (const float*)d_in[30];

  char* ws = (char*)d_ws;
  __bf16* xbf    = (__bf16*)(ws + O_XBF);
  __bf16* xbbf   = (__bf16*)(ws + O_XBBF);
  __bf16* w1f_t  = (__bf16*)(ws + O_W1F_T);
  __bf16* w1b_t  = (__bf16*)(ws + O_W1B_T);
  __bf16* w2f_t  = (__bf16*)(ws + O_W2F_T);
  __bf16* w2b_t  = (__bf16*)(ws + O_W2B_T);
  __bf16* whht   = (__bf16*)(ws + O_WHHT);
  float*  bias4  = (float*)(ws + O_BIAS4);
  __bf16* wtagt  = (__bf16*)(ws + O_WTAGT);
  float*  btag16 = (float*)(ws + O_BTAG);
  __bf16* wbint  = (__bf16*)(ws + O_WBINT);
  float*  bbin16 = (float*)(ws + O_BBIN);
  float*  zin    = (float*)(ws + O_ZIN);
  __bf16* finalbf = (__bf16*)(ws + O_FINAL);
  float*  emis   = (float*)(ws + O_EMIS);
  float*  emisb  = (float*)(ws + O_EMISB);
  __bf16* hbuf   = (__bf16*)(ws + O_HBUF);
  unsigned* ctr  = (unsigned*)(ws + O_CTR);
  float* out = (float*)d_out;

  // ---- phase 1: conversions / transposes / bias folds ----
  k_f2bf<<<8192, 256, 0, stream>>>(x,  xbf,  (size_t)BS * EMB);
  k_f2bf<<<16384, 256, 0, stream>>>(xb, xbbf, (size_t)BS * 3 * EMB);
  k_transpose_bf<<<2048, 256, 0, stream>>>(w1f_ih, w1f_t, GATES, EMB);
  k_transpose_bf<<<2048, 256, 0, stream>>>(w1b_ih, w1b_t, GATES, EMB);
  k_transpose_bf<<<6144, 256, 0, stream>>>(w2f_ih, w2f_t, GATES, 3 * EMB);
  k_transpose_bf<<<6144, 256, 0, stream>>>(w2b_ih, w2b_t, GATES, 3 * EMB);
  k_transpose_bf<<<1024, 256, 0, stream>>>(w1f_hh, whht + 0 * (size_t)HID * GATES, GATES, HID);
  k_transpose_bf<<<1024, 256, 0, stream>>>(w1b_hh, whht + 1 * (size_t)HID * GATES, GATES, HID);
  k_transpose_bf<<<1024, 256, 0, stream>>>(w2f_hh, whht + 2 * (size_t)HID * GATES, GATES, HID);
  k_transpose_bf<<<1024, 256, 0, stream>>>(w2b_hh, whht + 3 * (size_t)HID * GATES, GATES, HID);
  k_transpose_bf<<<64, 256, 0, stream>>>(w_tag, wtagt, 16, GATES);
  k_transpose_pad_bf<<<32, 256, 0, stream>>>(w_bin, wbint, 5, 2 * HID, 16);
  k_bias_add<<<4, 256, 0, stream>>>(b1f_ih, b1f_hh, bias4 + 0 * GATES, GATES);
  k_bias_add<<<4, 256, 0, stream>>>(b1b_ih, b1b_hh, bias4 + 1 * GATES, GATES);
  k_bias_add<<<4, 256, 0, stream>>>(b2f_ih, b2f_hh, bias4 + 2 * GATES, GATES);
  k_bias_add<<<4, 256, 0, stream>>>(b2b_ih, b2b_hh, bias4 + 3 * GATES, GATES);
  k_copy_pad<<<1, 32, 0, stream>>>(b_tag, btag16, 16, 16);
  k_copy_pad<<<1, 32, 0, stream>>>(bb_bin, bbin16, 5, 16);
  k_zero_u32<<<1, 32, 0, stream>>>(ctr, 8);

  // ---- phase 2: time-parallel input projections (Zin = x @ wih.T + bias) ----
  const size_t gemm_lds = 128 * 32 * 2 + 8 * 512 * 2;  // 16384
  dim3 gproj(BS / 128, GATES / 128);
  k_gemm_bf16<<<gproj, 256, gemm_lds, stream>>>(
      xbf, EMB, w1f_t, GATES, bias4 + 0 * GATES,
      zin + 0 * (size_t)BS * GATES, GATES, BS, GATES, EMB);
  k_gemm_bf16<<<gproj, 256, gemm_lds, stream>>>(
      xbf, EMB, w1b_t, GATES, bias4 + 1 * GATES,
      zin + 1 * (size_t)BS * GATES, GATES, BS, GATES, EMB);
  k_gemm_bf16<<<gproj, 256, gemm_lds, stream>>>(
      xbbf, 3 * EMB, w2f_t, GATES, bias4 + 2 * GATES,
      zin + 2 * (size_t)BS * GATES, GATES, BS, GATES, 3 * EMB);
  k_gemm_bf16<<<gproj, 256, gemm_lds, stream>>>(
      xbbf, 3 * EMB, w2b_t, GATES, bias4 + 3 * GATES,
      zin + 3 * (size_t)BS * GATES, GATES, BS, GATES, 3 * EMB);

  // ---- phase 3: 4 bidirectional LSTM recurrences ----
  const size_t rec_lds = 256 * 128 * 2 + 32 * 256 * 2 + 32 * 128 * 4 + 32 * 32 * 4; // 102400
  k_recurrent<<<32, 256, rec_lds, stream>>>(
      zin, whht, h0_main, c0_main, h0_bin, c0_bin, finalbf, hbuf, ctr);

  // ---- phase 4: emissions + CRF ----
  k_gemm_bf16<<<dim3(BS / 128, 1), 256, gemm_lds, stream>>>(
      finalbf, GATES, wtagt, 16, btag16, emis, 16, BS, 16, GATES);
  k_gemm_bf16<<<dim3(BS / 128, 1), 256, gemm_lds, stream>>>(
      finalbf + 2 * HID, GATES, wbint, 16, bbin16, emisb, 16, BS, 16, 2 * HID);

  k_crf<<<1, 512, 0, stream>>>(emis, 16, trans, 16, tags, mask, out, 0);
  k_crf<<<1, 512, 0, stream>>>(emisb, 16, transb, 5, tagsb, mask, out, 1);
}
